// GraphDiTBlock_1022202217267
// MI455X (gfx1250) — matrix operand order repaired and checked by reading, not compiled, not run on hardware
//
#include <hip/hip_runtime.h>
#include <hip/hip_bf16.h>

// ---------------- constants (match reference) ----------------
#define TN   8192      // N*K tokens
#define CDIM 128       // channels
#define HH   4         // heads
#define DD   32        // head dim
#define MODN 256       // time-encoding width
#define CTXF 7         // context features
#define FF   512       // ffn hidden

typedef __attribute__((ext_vector_type(16))) _Float16 v16h;
typedef __attribute__((ext_vector_type(8)))  _Float16 v8h;
typedef __attribute__((ext_vector_type(8)))  float    v8f;

// ---------------------------------------------------------------------------
// WMMA tile helper: 16x16 output tile of A(16xK) @ W^T.
// W is (Cout, Cin) row-major f16; wave owns columns [colbase, colbase+16).
// A is f16 row-major with stride lda (LDS).  v_wmma_f32_16x16x32_f16,
// f32 accumulate.  A 16-bit layout: lane m = row, g = lane>>4,
// halves[e] = K kb+16g+e (contiguous, same as B); B: lane n = column,
// halves[e] = W[colbase+n][kb+16g+e] -> contiguous 32B load.
// ---------------------------------------------------------------------------
__device__ inline v8f wmma_f16_tile(const _Float16* __restrict__ A, int lda,
                                    const _Float16* __restrict__ W, int ldw,
                                    int colbase, int Kdim, v8f acc) {
  const int lane = threadIdx.x & 31;
  const int m = lane & 15;        // row for A, column for B
  const int g = lane >> 4;
  for (int kb = 0; kb < Kdim; kb += 32) {
    const v8h* pa = (const v8h*)(A + m * lda + kb + g * 16);
    v8h alo = pa[0];
    v8h ahi = pa[1];            // +8 halves
    v16h av;
#pragma unroll
    for (int e = 0; e < 8; ++e) { av[e] = alo[e]; av[e + 8] = ahi[e]; }
    v16h bv = *(const v16h*)(W + (size_t)(colbase + m) * ldw + kb + 16 * g);
    acc = __builtin_amdgcn_wmma_f32_16x16x32_f16(
        /*neg_a=*/false, av, /*neg_b=*/false, bv,
        /*c_mod=*/(short)0, acc, /*reuse_a=*/false, /*reuse_b=*/false);
  }
  return acc;
}

__device__ inline void atomicMaxFloat(float* addr, float val) {
  unsigned int* ua = (unsigned int*)addr;
  unsigned int old = *ua;
  while (true) {
    float f = __uint_as_float(old);
    if (f >= val) break;
    unsigned int assumed = old;
    old = atomicCAS(ua, assumed, __float_as_uint(val));
    if (old == assumed) break;
  }
}

__device__ inline float silu_f(float z) { return z / (1.f + __expf(-z)); }

// ---------------------------------------------------------------------------
// 1) time modulation MLP: mod[384] = silu(te @ w1^T + b1) @ w2^T + b2
// ---------------------------------------------------------------------------
__global__ void mod_kernel(const float* __restrict__ te,
                           const float* __restrict__ w1, const float* __restrict__ b1,
                           const float* __restrict__ w2, const float* __restrict__ b2,
                           float* __restrict__ mod) {
  __shared__ float st[MODN];
  const int tid = threadIdx.x;
  {
    float acc = b1[tid];
    for (int j = 0; j < MODN; ++j) acc += te[j] * w1[tid * MODN + j];
    st[tid] = silu_f(acc);
  }
  __syncthreads();
  for (int o = tid; o < 3 * CDIM; o += MODN) {
    float acc = b2[o];
    for (int j = 0; j < MODN; ++j) acc += st[j] * w2[o * MODN + j];
    mod[o] = acc;
  }
}

// ---------------------------------------------------------------------------
// 2) f32 -> f16 weight conversion
// ---------------------------------------------------------------------------
__global__ void f32_to_f16_kernel(const float* __restrict__ src,
                                  _Float16* __restrict__ dst, int n) {
  int i = blockIdx.x * blockDim.x + threadIdx.x;
  if (i < n) dst[i] = (_Float16)src[i];
}

// ---------------------------------------------------------------------------
// 3) FUSED FRONT: siren ctx -> adaLN -> y ; then Q/K/V (WMMA) + per-head LN.
//    Per 16-token tile; y kept in LDS between stages; writes y32,q,k,v.
// ---------------------------------------------------------------------------
__global__ void fused_front_kernel(const float* __restrict__ x,
                                   const float* __restrict__ ctx_enc,
                                   const float* __restrict__ sw0, const float* __restrict__ sb0,
                                   const _Float16* __restrict__ sw1_16, const float* __restrict__ sb1,
                                   const float* __restrict__ mod,
                                   const _Float16* __restrict__ wq16, const float* __restrict__ bq,
                                   const _Float16* __restrict__ wk16, const float* __restrict__ bk,
                                   const _Float16* __restrict__ wv16, const float* __restrict__ bv,
                                   float* __restrict__ y32,
                                   float* __restrict__ qo, float* __restrict__ ko,
                                   float* __restrict__ vo) {
  __shared__ _Float16 sh[16 * CDIM];     // hctx (f16 WMMA operand)
  __shared__ float    stmp[16 * CDIM];   // x + ctx (pre-LN)
  __shared__ _Float16 sy[16 * CDIM];     // y (f16 WMMA operand)
  __shared__ float    so[16 * CDIM];     // q/k staging for per-head LN
  const int row0 = blockIdx.x * 16;
  const int tid = threadIdx.x;

  // phase 1: hctx = sin(ctx_enc @ sw0^T + sb0)  (K=7, VALU)
  for (int idx = tid; idx < 16 * CDIM; idx += 256) {
    int r = idx >> 7, cc = idx & 127;
    const float* ce = ctx_enc + (size_t)(row0 + r) * CTXF;
    float acc = sb0[cc];
#pragma unroll
    for (int f = 0; f < CTXF; ++f) acc += ce[f] * sw0[cc * CTXF + f];
    sh[idx] = (_Float16)__sinf(acc);
  }
  __syncthreads();

  // phase 2: ctx = hctx @ sw1^T + sb1 ; add x  (WMMA)
  {
    const int w = tid >> 5, lane = tid & 31;
    const int colbase = w * 16;
    v8f acc = {};
    acc = wmma_f16_tile(sh, CDIM, sw1_16, CDIM, colbase, CDIM, acc);
    const int n = lane & 15, g = lane >> 4;
#pragma unroll
    for (int i = 0; i < 8; ++i) {
      int r = i + 8 * g, cc = colbase + n;
      stmp[r * CDIM + cc] = acc[i] + sb1[cc] + x[(size_t)(row0 + r) * CDIM + cc];
    }
  }
  __syncthreads();

  // phase 3: row LN (16 lanes/row, shuffle tree) + adaLN -> sy, y32
  {
    const int r = tid >> 4, l = tid & 15;   // 16 threads per row, same half-wave
    float s = 0.f, sq = 0.f;
#pragma unroll
    for (int j = 0; j < 8; ++j) {
      float v2 = stmp[r * CDIM + l + 16 * j];
      s += v2; sq += v2 * v2;
    }
#pragma unroll
    for (int mk = 1; mk < 16; mk <<= 1) {
      s += __shfl_xor(s, mk, 16);
      sq += __shfl_xor(sq, mk, 16);
    }
    float mu = s * (1.f / CDIM);
    float var = sq * (1.f / CDIM) - mu * mu;
    float rs = rsqrtf(fmaxf(var, 0.f) + 1e-5f);
#pragma unroll
    for (int j = 0; j < 8; ++j) {
      int cc = l + 16 * j;
      float z = (stmp[r * CDIM + cc] - mu) * rs;
      float yv = (mod[cc] + 1.f) * z + mod[CDIM + cc];
      sy[r * CDIM + cc] = (_Float16)yv;
      y32[(size_t)(row0 + r) * CDIM + cc] = yv;
    }
  }
  __syncthreads();

  // phase 4: Q/K/V projections (WMMA); q,k get per-(token,head) LN over 32
  const int w = tid >> 5, lane = tid & 31;
  const int colbase = w * 16, n = lane & 15, g = lane >> 4;
  for (int pi = 0; pi < 3; ++pi) {
    const _Float16* Wt = (pi == 0) ? wq16 : (pi == 1) ? wk16 : wv16;
    const float* bias  = (pi == 0) ? bq   : (pi == 1) ? bk   : bv;
    float* outp        = (pi == 0) ? qo   : (pi == 1) ? ko   : vo;
    v8f acc = {};
    acc = wmma_f16_tile(sy, CDIM, Wt, CDIM, colbase, CDIM, acc);
    if (pi == 2) {  // v: no LN, write straight from accumulators
#pragma unroll
      for (int i = 0; i < 8; ++i) {
        int r = i + 8 * g, cc = colbase + n;
        outp[(size_t)(row0 + r) * CDIM + cc] = acc[i] + bias[cc];
      }
    } else {
#pragma unroll
      for (int i = 0; i < 8; ++i) {
        int r = i + 8 * g, cc = colbase + n;
        so[r * CDIM + cc] = acc[i] + bias[cc];
      }
      __syncthreads();
      // per-(row, head) LN over D=32: 4 lanes/group, 8 elems each, shuffle tree
      {
        const int grp = tid >> 2;          // 64 groups (row, head)
        const int r = grp >> 2, h = grp & 3;
        const int sub = tid & 3;
        const float* pbase = so + r * CDIM + h * DD;
        float s = 0.f, sq = 0.f;
#pragma unroll
        for (int j = 0; j < 8; ++j) {
          float v2 = pbase[sub * 8 + j];
          s += v2; sq += v2 * v2;
        }
#pragma unroll
        for (int mk = 1; mk < 4; mk <<= 1) {
          s += __shfl_xor(s, mk, 4);
          sq += __shfl_xor(sq, mk, 4);
        }
        float mu = s * (1.f / DD);
        float var = sq * (1.f / DD) - mu * mu;
        float rs = rsqrtf(fmaxf(var, 0.f) + 1e-5f);
#pragma unroll
        for (int j = 0; j < 8; ++j) {
          int d = sub * 8 + j;
          outp[(size_t)(row0 + r) * CDIM + h * DD + d] = (pbase[d] - mu) * rs;
        }
      }
      __syncthreads();   // before so is overwritten by next projection
    }
  }
}

// ---------------------------------------------------------------------------
// 4) edge softmax (scatter by src, L2 atomics)
// ---------------------------------------------------------------------------
__global__ void attn_init_kernel(float* __restrict__ m, float* __restrict__ denom,
                                 float* __restrict__ agg, int nm, int nagg) {
  int i = blockIdx.x * blockDim.x + threadIdx.x;
  if (i < nm) { m[i] = -1e30f; denom[i] = 0.f; }
  if (i < nagg) agg[i] = 0.f;
}

__global__ void edge_score_kernel(const int* __restrict__ edges,
                                  const float* __restrict__ q, const float* __restrict__ k,
                                  const float* __restrict__ v,
                                  float* __restrict__ s, float* __restrict__ m, int E) {
  int gid = blockIdx.x * blockDim.x + threadIdx.x;
  if (gid >= E * HH) return;
  int e = gid >> 2, h = gid & 3;
  int src = edges[2 * e], dst = edges[2 * e + 1];
  const float* qp = q + (size_t)src * CDIM + h * DD;
  const float* kp = k + (size_t)dst * CDIM + h * DD;
  // warm L2 with the v row the accumulate pass will gather (global_prefetch_b8)
  __builtin_prefetch(v + (size_t)dst * CDIM + h * DD, 0, 1);
  float acc = 0.f;
#pragma unroll
  for (int d = 0; d < DD; ++d) acc += qp[d] * kp[d];
  acc *= 0.17677669529663687f;  // 1/sqrt(32)
  s[gid] = acc;
  atomicMaxFloat(&m[src * HH + h], acc);
}

__global__ void edge_accum_kernel(const int* __restrict__ edges,
                                  const float* __restrict__ v, const float* __restrict__ s,
                                  const float* __restrict__ m, float* __restrict__ denom,
                                  float* __restrict__ agg, int E) {
  int gid = blockIdx.x * blockDim.x + threadIdx.x;
  if (gid >= E * HH) return;
  int e = gid >> 2, h = gid & 3;
  int src = edges[2 * e], dst = edges[2 * e + 1];
  float p = __expf(s[gid] - m[src * HH + h]);
  atomicAdd(&denom[src * HH + h], p);
  const float* vp = v + (size_t)dst * CDIM + h * DD;
  float* ap = agg + (size_t)src * CDIM + h * DD;
#pragma unroll 8
  for (int d = 0; d < DD; ++d) atomicAdd(&ap[d], p * vp[d]);
}

// ---------------------------------------------------------------------------
// 5) FUSED TAIL: attn normalize -> Wo + residual -> FFN1 -> FFN2 -> gate.
//    attn / y2 / h1 live only in LDS (24 KB/block); writes final output.
// ---------------------------------------------------------------------------
__global__ void fused_tail_kernel(const float* __restrict__ agg,
                                  const float* __restrict__ denom,
                                  const _Float16* __restrict__ wo16, const float* __restrict__ bo,
                                  const _Float16* __restrict__ fw1_16, const float* __restrict__ fb1,
                                  const _Float16* __restrict__ fw2_16, const float* __restrict__ fb2,
                                  const float* __restrict__ y32,
                                  const float* __restrict__ x,
                                  const float* __restrict__ mod,
                                  float* __restrict__ out) {
  __shared__ _Float16 sattn[16 * CDIM];  // normalized attention (f16)
  __shared__ _Float16 sy2[16 * CDIM];    // y + attn@wo^T + bo (f16)
  __shared__ _Float16 sh1[16 * FF];      // silu(ffn1) (f16)
  const int row0 = blockIdx.x * 16;
  const int tid = threadIdx.x;

  for (int idx = tid; idx < 16 * CDIM; idx += 256) {
    int r = idx >> 7, h = (idx & 127) >> 5;
    float den = fmaxf(denom[(row0 + r) * HH + h], 1e-9f);
    sattn[idx] = (_Float16)(agg[(size_t)row0 * CDIM + idx] / den);
  }
  __syncthreads();

  const int w = tid >> 5, lane = tid & 31;
  const int colbase = w * 16, n = lane & 15, g = lane >> 4;

  // Wo + residual -> sy2 (LDS only)
  {
    v8f acc = {};
    acc = wmma_f16_tile(sattn, CDIM, wo16, CDIM, colbase, CDIM, acc);
#pragma unroll
    for (int i = 0; i < 8; ++i) {
      int r = i + 8 * g, cc = colbase + n;
      float yv = y32[(size_t)(row0 + r) * CDIM + cc] + acc[i] + bo[cc];
      sy2[r * CDIM + cc] = (_Float16)yv;
    }
  }
  __syncthreads();

  // FFN1: 128 -> 512, silu  (8 waves x 4 column tiles)
  for (int t = 0; t < 4; ++t) {
    int cb = (w * 4 + t) * 16;
    v8f acc = {};
    acc = wmma_f16_tile(sy2, CDIM, fw1_16, CDIM, cb, CDIM, acc);
#pragma unroll
    for (int i = 0; i < 8; ++i) {
      int r = i + 8 * g, cc = cb + n;
      sh1[r * FF + cc] = (_Float16)silu_f(acc[i] + fb1[cc]);
    }
  }
  __syncthreads();

  // FFN2 (K=512) + final gate: out = (x + c*ffn) * rsqrt(1 + c*c)
  {
    v8f acc = {};
    acc = wmma_f16_tile(sh1, FF, fw2_16, FF, colbase, FF, acc);
#pragma unroll
    for (int i = 0; i < 8; ++i) {
      int r = i + 8 * g, cc = colbase + n;
      float yv = acc[i] + fb2[cc];
      float cv = mod[2 * CDIM + cc];
      size_t o = (size_t)(row0 + r) * CDIM + cc;
      out[o] = (x[o] + cv * yv) * rsqrtf(1.f + cv * cv);
    }
  }
}

// ---------------------------------------------------------------------------
// launcher
// ---------------------------------------------------------------------------
extern "C" void kernel_launch(void* const* d_in, const int* in_sizes, int n_in,
                              void* d_out, int out_size, void* d_ws, size_t ws_size,
                              hipStream_t stream) {
  (void)n_in; (void)out_size; (void)ws_size;
  const float* x        = (const float*)d_in[0];
  const float* te       = (const float*)d_in[1];
  const float* ctx_enc  = (const float*)d_in[2];
  const int*   edges    = (const int*)d_in[3];
  const float* mod_w1   = (const float*)d_in[4];
  const float* mod_b1   = (const float*)d_in[5];
  const float* mod_w2   = (const float*)d_in[6];
  const float* mod_b2   = (const float*)d_in[7];
  const float* siren_w0 = (const float*)d_in[8];
  const float* siren_b0 = (const float*)d_in[9];
  const float* siren_w1 = (const float*)d_in[10];
  const float* siren_b1 = (const float*)d_in[11];
  const float* wq = (const float*)d_in[12]; const float* bq = (const float*)d_in[13];
  const float* wk = (const float*)d_in[14]; const float* bk = (const float*)d_in[15];
  const float* wv = (const float*)d_in[16]; const float* bv = (const float*)d_in[17];
  const float* wo = (const float*)d_in[18]; const float* bo = (const float*)d_in[19];
  const float* fw1 = (const float*)d_in[20]; const float* fb1 = (const float*)d_in[21];
  const float* fw2 = (const float*)d_in[22]; const float* fb2 = (const float*)d_in[23];
  float* out = (float*)d_out;

  const int E = in_sizes[3] / 2;

  // ---- workspace carve-out (256B aligned) ----
  char* p = (char*)d_ws;
  auto alloc = [&](size_t bytes) -> void* {
    void* r = (void*)p;
    p += (bytes + 255) & ~(size_t)255;
    return r;
  };
  float*     mod     = (float*)alloc(3 * CDIM * sizeof(float));
  _Float16*  wq16    = (_Float16*)alloc(CDIM * CDIM * 2);
  _Float16*  wk16    = (_Float16*)alloc(CDIM * CDIM * 2);
  _Float16*  wv16    = (_Float16*)alloc(CDIM * CDIM * 2);
  _Float16*  wo16    = (_Float16*)alloc(CDIM * CDIM * 2);
  _Float16*  sw1_16  = (_Float16*)alloc(CDIM * CDIM * 2);
  _Float16*  fw1_16  = (_Float16*)alloc(FF * CDIM * 2);
  _Float16*  fw2_16  = (_Float16*)alloc(CDIM * FF * 2);
  float*     y32     = (float*)alloc((size_t)TN * CDIM * 4);
  float*     q32     = (float*)alloc((size_t)TN * CDIM * 4);
  float*     k32     = (float*)alloc((size_t)TN * CDIM * 4);
  float*     v32     = (float*)alloc((size_t)TN * CDIM * 4);
  float*     s_ws    = (float*)alloc((size_t)E * HH * 4);
  float*     m_ws    = (float*)alloc((size_t)TN * HH * 4);
  float*     den_ws  = (float*)alloc((size_t)TN * HH * 4);
  float*     agg_ws  = (float*)alloc((size_t)TN * CDIM * 4);

  const int tiles = TN / 16;  // 512 row-tiles, 256 threads (8 waves) each

  // 1) modulation MLP
  mod_kernel<<<1, MODN, 0, stream>>>(te, mod_w1, mod_b1, mod_w2, mod_b2, mod);

  // 2) weight conversion to f16
  auto cvt = [&](const float* s, _Float16* d, int n) {
    f32_to_f16_kernel<<<(n + 255) / 256, 256, 0, stream>>>(s, d, n);
  };
  cvt(wq, wq16, CDIM * CDIM);
  cvt(wk, wk16, CDIM * CDIM);
  cvt(wv, wv16, CDIM * CDIM);
  cvt(wo, wo16, CDIM * CDIM);
  cvt(siren_w1, sw1_16, CDIM * CDIM);
  cvt(fw1, fw1_16, FF * CDIM);
  cvt(fw2, fw2_16, CDIM * FF);

  // 3) fused context + adaLN + QKV (+per-head LN)
  fused_front_kernel<<<tiles, 256, 0, stream>>>(
      x, ctx_enc, siren_w0, siren_b0, sw1_16, siren_b1, mod,
      wq16, bq, wk16, bk, wv16, bv, y32, q32, k32, v32);

  // 4) graph attention (edge softmax via L2 atomics)
  {
    int nagg = TN * CDIM, nm = TN * HH;
    attn_init_kernel<<<(nagg + 255) / 256, 256, 0, stream>>>(m_ws, den_ws, agg_ws, nm, nagg);
    int nedge = E * HH;
    edge_score_kernel<<<(nedge + 255) / 256, 256, 0, stream>>>(edges, q32, k32, v32,
                                                               s_ws, m_ws, E);
    edge_accum_kernel<<<(nedge + 255) / 256, 256, 0, stream>>>(edges, v32, s_ws, m_ws,
                                                               den_ws, agg_ws, E);
  }

  // 5) fused attn-normalize + Wo + residual + FFN + final gate
  fused_tail_kernel<<<tiles, 256, 0, stream>>>(
      agg_ws, den_ws, wo16, bo, fw1_16, fb1, fw2_16, fb2, y32, x, mod, out);
}
